// EmformerAttention_41257455845768
// MI455X (gfx1250) — matrix operand order repaired and checked by reading, not compiled, not run on hardware
//
#include <hip/hip_runtime.h>

// ---------------- problem constants ----------------
#define EMBED 512
#define NHEAD 8
#define HEADD 64
#define Bsz 16
#define Uu 1024
#define Rr 256
#define Ss 32
#define Mm 31
#define Qq 1312            // R+U+S
#define KVv 1311           // M+R+U
#define KVP 1312           // padded KV (multiple of 32)
#define NEG_INF -100000000.0f

#define HB (Bsz*NHEAD)       // 128
#define HSTRIDE (KVP*HEADD)  // 83968 halves per head-batch
#define QH_E ((size_t)HB*HSTRIDE)
#define NKVSTEPS (KVP/32)    // 41

typedef __attribute__((ext_vector_type(16))) _Float16 v16h_t;
typedef __attribute__((ext_vector_type(8)))  _Float16 v8h_t;
typedef __attribute__((ext_vector_type(8)))  float    v8f_t;

union V16U { v16h_t v; v8h_t h2[2]; _Float16 h[16]; };

// ---- WMMA wrapper: D = A(16x32 f16) * B(32x16 f16) + C(16x16 f32) ----
__device__ __forceinline__ v8f_t wmma_f16(v16h_t a, v16h_t b, v8f_t c) {
  return __builtin_amdgcn_wmma_f32_16x16x32_f16(false, a, false, b, (short)0, c,
                                                false, false);
}

// A-matrix 16x32 f16 layout: lane L -> m = L&15; kb = (L>=16)?8:0
__device__ __forceinline__ v16h_t load_a_f32(const float* row, int kstep, int lane) {
  int kb = (lane >> 4) << 3;
  const float* p0 = row + kstep + kb;
  const float* p1 = p0 + 16;
  V16U u;
#pragma unroll
  for (int i = 0; i < 8; ++i) { u.h[i] = (_Float16)p0[i]; u.h[8+i] = (_Float16)p1[i]; }
  return u.v;
}
__device__ __forceinline__ v16h_t load_a_f16(const _Float16* row, int kstep, int lane) {
  int kb = (lane >> 4) << 3;
  V16U u;
  u.h2[0] = *(const v8h_t*)(row + kstep + kb);
  u.h2[1] = *(const v8h_t*)(row + kstep + kb + 16);
  return u.v;
}

// B-matrix 32x16 f16 layout: lane L -> n = L&15; kb2 = (L>=16)?8:0
__device__ __forceinline__ v16h_t load_b_f32(const float* colk, int kstep, int lane) {
  int kb2 = (lane >> 4) << 3;
  const float* p = colk + kstep + kb2;
  V16U u;
#pragma unroll
  for (int i = 0; i < 8; ++i) { u.h[i] = (_Float16)p[i]; u.h[8+i] = (_Float16)p[16+i]; }
  return u.v;
}
__device__ __forceinline__ v16h_t load_b_h16(const _Float16* colk, int lane) {
  int kb2 = (lane >> 4) << 3;
  V16U u;
  u.h2[0] = *(const v8h_t*)(colk + kb2);
  u.h2[1] = *(const v8h_t*)(colk + kb2 + 16);
  return u.v;
}

// ---- CDNA5 async global->LDS copy (ASYNCcnt-tracked) ----
__device__ __forceinline__ void async_b128(unsigned int lds_off, const void* gptr) {
  asm volatile("global_load_async_to_lds_b128 %0, %1, off"
               :: "v"(lds_off), "v"(gptr) : "memory");
}
__device__ __forceinline__ void wait_async4() {
  asm volatile("s_wait_asynccnt 0x4" ::: "memory");
}
__device__ __forceinline__ void wait_async0() {
  asm volatile("s_wait_asynccnt 0x0" ::: "memory");
}
__device__ __forceinline__ unsigned int lds_off32(const void* p) {
  return (unsigned int)(size_t)p;   // LDS aperture: addr[31:0] = workgroup LDS offset
}

// ============== kernel 1a: query projection ==============
__global__ void qproj_kernel(const float* __restrict__ utt, const float* __restrict__ rc,
                             const float* __restrict__ summ, const float* __restrict__ Wq,
                             const float* __restrict__ bq, _Float16* __restrict__ qh) {
  int wave = (blockIdx.x * blockDim.x + threadIdx.x) >> 5;
  int lane = threadIdx.x & 31;
  int mt  = wave >> 3;
  int n64 = (wave & 7) << 6;
  int col = lane & 15;

  int r = mt * 16 + col;
  const float* arow = (r < Rr*Bsz)      ? rc  + (size_t)r * EMBED
                    : (r < (Rr+Uu)*Bsz) ? utt + (size_t)(r - Rr*Bsz) * EMBED
                                        : summ + (size_t)(r - (Rr+Uu)*Bsz) * EMBED;

  v8f_t acc[4] = {v8f_t{}, v8f_t{}, v8f_t{}, v8f_t{}};
  for (int kstep = 0; kstep < EMBED; kstep += 32) {
    v16h_t A = load_a_f32(arow, kstep, lane);
#pragma unroll
    for (int nn = 0; nn < 4; ++nn) {
      int n = n64 + nn * 16 + col;
      acc[nn] = wmma_f16(A, load_b_f32(Wq + (size_t)n * EMBED, kstep, lane), acc[nn]);
    }
  }
  int half = lane >> 4;
#pragma unroll
  for (int nn = 0; nn < 4; ++nn) {
    int c = n64 + nn * 16 + col;
    int h = c >> 6, d = c & 63;
    float bias = bq[c];
#pragma unroll
    for (int j = 0; j < 8; ++j) {
      int bb = j + (half << 3);
      int hb = bb * NHEAD + h;
      float v = (acc[nn][j] + bias) * 0.125f;
      qh[(size_t)hb * HSTRIDE + (size_t)mt * HEADD + d] = (_Float16)v;
    }
  }
}

// ============== kernel 1b: key/value projection ==============
__global__ void kvproj_kernel(const float* __restrict__ utt, const float* __restrict__ rc,
                              const float* __restrict__ mem, const float* __restrict__ Wkv,
                              const float* __restrict__ bkv, _Float16* __restrict__ kh,
                              _Float16* __restrict__ vhT) {
  int wave = (blockIdx.x * blockDim.x + threadIdx.x) >> 5;
  int lane = threadIdx.x & 31;
  int mt  = wave >> 4;
  int n64 = (wave & 15) << 6;
  int col = lane & 15;

  int r = mt * 16 + col;
  const float* arow = (r < Mm*Bsz)      ? mem + (size_t)r * EMBED
                    : (r < (Mm+Rr)*Bsz) ? rc  + (size_t)(r - Mm*Bsz) * EMBED
                                        : utt + (size_t)(r - (Mm+Rr)*Bsz) * EMBED;

  v8f_t acc[4] = {v8f_t{}, v8f_t{}, v8f_t{}, v8f_t{}};
  for (int kstep = 0; kstep < EMBED; kstep += 32) {
    v16h_t A = load_a_f32(arow, kstep, lane);
#pragma unroll
    for (int nn = 0; nn < 4; ++nn) {
      int n = n64 + nn * 16 + col;
      acc[nn] = wmma_f16(A, load_b_f32(Wkv + (size_t)n * EMBED, kstep, lane), acc[nn]);
    }
  }
  int half = lane >> 4;
#pragma unroll
  for (int nn = 0; nn < 4; ++nn) {
    int c = n64 + nn * 16 + col;
    float bias = bkv[c];
#pragma unroll
    for (int j = 0; j < 8; ++j) {
      int bb = j + (half << 3);
      float v = acc[nn][j] + bias;
      if (c < EMBED) {
        int h = c >> 6, d = c & 63;
        int hb = bb * NHEAD + h;
        kh[(size_t)hb * HSTRIDE + (size_t)mt * HEADD + d] = (_Float16)v;
      } else {
        int c2 = c - EMBED;
        int h = c2 >> 6, d = c2 & 63;
        int hb = bb * NHEAD + h;
        vhT[(size_t)hb * HSTRIDE + (size_t)d * KVP + mt] = (_Float16)v;
      }
    }
  }
}

// ============== kernel 1c: zero kv pad slot (t = 1311) ==============
__global__ void pad_kernel(_Float16* __restrict__ kh, _Float16* __restrict__ vhT) {
  int i = blockIdx.x * blockDim.x + threadIdx.x;
  if (i < HB * HEADD) {
    int hb = i >> 6, d = i & 63;
    kh [(size_t)hb * HSTRIDE + (size_t)(KVP - 1) * HEADD + d] = (_Float16)0.0f;
    vhT[(size_t)hb * HSTRIDE + (size_t)d * KVP + (KVP - 1)]  = (_Float16)0.0f;
  }
}

// ============== kernel 2: flash attention ==============
// Workgroup = 4 waves sharing one head-batch hb; wave wv handles q-tile qg*4+wv.
// K/V tiles (32 kv cols) staged in LDS via async copies, double-buffered.
__global__ void attn_kernel(const unsigned char* __restrict__ amask,
                            const unsigned char* __restrict__ pmask,
                            const _Float16* __restrict__ qh, const _Float16* __restrict__ kh,
                            const _Float16* __restrict__ vhT, _Float16* __restrict__ attnH) {
  __shared__ __align__(16) _Float16 kbuf[2][32][64];  // [buf][kv-local][d]
  __shared__ __align__(16) _Float16 vbuf[2][64][32];  // [buf][d][kv-local]
  __shared__ __align__(16) _Float16 pbuf[4][16][40];  // P tile per wave

  int tid  = threadIdx.x;
  int wv   = tid >> 5;
  int lane = tid & 31;
  int hb = blockIdx.x & (HB - 1);       // 0..127
  int qg = blockIdx.x >> 7;             // 0..20
  int qslot = qg * 4 + wv;
  bool active = qslot < (Qq / 16);
  int qt = active ? qslot : (Qq / 16 - 1);
  int bidx = hb >> 3, hh = hb & 7;
  size_t hoff = (size_t)hb * HSTRIDE;
  int col = lane & 15;
  int half = lane >> 4;

  const _Float16* khb = kh + hoff;
  const _Float16* vhb = vhT + hoff;

  // Q A-tiles (16x64), reused for every kv step
  int qrow = qt * 16 + col;
  const _Float16* qrp = qh + hoff + (size_t)qrow * HEADD;
  v16h_t A0 = load_a_f16(qrp, 0, lane);
  v16h_t A1 = load_a_f16(qrp, 32, lane);

  float mst[8], lst[8];
#pragma unroll
  for (int j = 0; j < 8; ++j) { mst[j] = -3.0e38f; lst[j] = 0.0f; }
  v8f_t o[4] = {v8f_t{}, v8f_t{}, v8f_t{}, v8f_t{}};

  // ---- async tile copy: 4 x b128 per thread (K: 4KB contiguous, V: 64 rows x 64B)
  auto issue_tiles = [&](int kvb, int buf) {
    const char* ksrc = (const char*)(khb + (size_t)kvb * HEADD);
    unsigned int koff = lds_off32(&kbuf[buf][0][0]);
    async_b128(koff + tid * 16,        ksrc + tid * 16);
    async_b128(koff + 2048 + tid * 16, ksrc + 2048 + tid * 16);
    unsigned int voff = lds_off32(&vbuf[buf][0][0]);
    int ch0 = tid, ch1 = tid + 128;
    async_b128(voff + ch0 * 16, vhb + (size_t)(ch0 >> 2) * KVP + kvb + (ch0 & 3) * 8);
    async_b128(voff + ch1 * 16, vhb + (size_t)(ch1 >> 2) * KVP + kvb + (ch1 & 3) * 8);
  };

  issue_tiles(0, 0);   // prologue

  for (int s = 0; s < NKVSTEPS; ++s) {
    int kvb = s * 32;
    int cur = s & 1;
    if (s + 1 < NKVSTEPS) { issue_tiles(kvb + 32, (s + 1) & 1); wait_async4(); }
    else                  { wait_async0(); }
    __syncthreads();     // staged K/V tile visible to all waves

    // ---- scores: two 16x16 tiles from LDS K ----
    int kvl0 = col, kvl1 = col + 16;
    v8f_t s0 = {}, s1 = {};
    s0 = wmma_f16(A0, load_b_h16(&kbuf[cur][kvl0][0],  lane), s0);
    s0 = wmma_f16(A1, load_b_h16(&kbuf[cur][kvl0][32], lane), s0);
    s1 = wmma_f16(A0, load_b_h16(&kbuf[cur][kvl1][0],  lane), s1);
    s1 = wmma_f16(A1, load_b_h16(&kbuf[cur][kvl1][32], lane), s1);

    // ---- masking + online softmax (kv index constant per lane) ----
    int kv0 = kvb + col, kv1 = kv0 + 16;
    bool dead0 = (kv0 >= KVv) || (pmask[bidx * KVv + kv0] != 0);
    bool dead1 = (kv1 >= KVv) || (pmask[bidx * KVv + kv1] != 0);
    float p0v[8], p1v[8], scl[8];
#pragma unroll
    for (int j = 0; j < 8; ++j) {
      int q = qt * 16 + j + (half << 3);
      float x0 = (dead0 || (amask[(size_t)q * KVv + kv0] != 0)) ? NEG_INF : s0[j];
      float x1 = (dead1 || (amask[(size_t)q * KVv + kv1] != 0)) ? NEG_INF : s1[j];
      float mx = fmaxf(x0, x1);
      for (int msk = 1; msk < 16; msk <<= 1)
        mx = fmaxf(mx, __shfl_xor(mx, msk, 32));
      float mnew = fmaxf(mst[j], mx);
      float sc = __expf(mst[j] - mnew);
      float e0 = __expf(x0 - mnew);
      float e1 = __expf(x1 - mnew);
      float rs = e0 + e1;
      for (int msk = 1; msk < 16; msk <<= 1)
        rs += __shfl_xor(rs, msk, 32);
      lst[j] = lst[j] * sc + rs;
      mst[j] = mnew;
      scl[j] = sc; p0v[j] = e0; p1v[j] = e1;
    }
#pragma unroll
    for (int nn = 0; nn < 4; ++nn)
#pragma unroll
      for (int j = 0; j < 8; ++j) o[nn][j] *= scl[j];

    // ---- P: C-layout -> LDS -> A-layout (16x32 f16) ----
    __syncthreads();
#pragma unroll
    for (int j = 0; j < 8; ++j) {
      int row = j + (half << 3);
      pbuf[wv][row][col]      = (_Float16)p0v[j];
      pbuf[wv][row][col + 16] = (_Float16)p1v[j];
    }
    __syncthreads();
    V16U pa;
    int kb = half << 3;
    pa.h2[0] = *(const v8h_t*)&pbuf[wv][col][kb];
    pa.h2[1] = *(const v8h_t*)&pbuf[wv][col][kb + 16];
    v16h_t P = pa.v;

    // ---- O += P(16x32) * V(32x64), V from LDS ----
#pragma unroll
    for (int nn = 0; nn < 4; ++nn)
      o[nn] = wmma_f16(P, load_b_h16(&vbuf[cur][nn * 16 + col][0], lane), o[nn]);

    __syncthreads();   // everyone done reading this buffer before it is refilled
  }

  // ---- finalize + store attn (token-row major f16 for the O-proj GEMM) ----
  if (active) {
    float inv[8];
#pragma unroll
    for (int j = 0; j < 8; ++j) inv[j] = 1.0f / lst[j];
#pragma unroll
    for (int nn = 0; nn < 4; ++nn) {
      int c = hh * HEADD + nn * 16 + col;
#pragma unroll
      for (int j = 0; j < 8; ++j) {
        int t = qt * 16 + j + (half << 3);
        size_t rr = (size_t)t * Bsz + bidx;
        attnH[rr * EMBED + c] = (_Float16)(o[nn][j] * inv[j]);
      }
    }
  }
}

// ============== kernel 3: output projection + clip/split ==============
__global__ void oproj_kernel(const _Float16* __restrict__ attnH, const float* __restrict__ Wo,
                             const float* __restrict__ bo, float* __restrict__ out) {
  int wave = (blockIdx.x * blockDim.x + threadIdx.x) >> 5;
  int lane = threadIdx.x & 31;
  int mt  = wave >> 3;
  int n64 = (wave & 7) << 6;
  int col = lane & 15;
  int half = lane >> 4;

  const _Float16* arow = attnH + ((size_t)mt * 16 + col) * EMBED;
  v8f_t acc[4] = {v8f_t{}, v8f_t{}, v8f_t{}, v8f_t{}};
  for (int kstep = 0; kstep < EMBED; kstep += 32) {
    v16h_t A = load_a_f16(arow, kstep, lane);
#pragma unroll
    for (int nn = 0; nn < 4; ++nn) {
      int n = n64 + nn * 16 + col;
      acc[nn] = wmma_f16(A, load_b_f32(Wo + (size_t)n * EMBED, kstep, lane), acc[nn]);
    }
  }
  int t = mt;
#pragma unroll
  for (int nn = 0; nn < 4; ++nn) {
    int n = n64 + nn * 16 + col;
    float bias = bo[n];
#pragma unroll
    for (int j = 0; j < 8; ++j) {
      int b = j + (half << 3);
      float v = acc[nn][j] + bias;
      if (t < Rr + Uu) {
        out[((size_t)t * Bsz + b) * EMBED + n] = v;
      } else if (t < Qq - 1) {
        v = fminf(fmaxf(v, -10.0f), 10.0f);
        out[(size_t)(Rr + Uu) * Bsz * EMBED +
            ((size_t)(t - (Rr + Uu)) * Bsz + b) * EMBED + n] = v;
      }
    }
  }
}

// ============== host launcher ==============
extern "C" void kernel_launch(void* const* d_in, const int* in_sizes, int n_in,
                              void* d_out, int out_size, void* d_ws, size_t ws_size,
                              hipStream_t stream) {
  const float* utt  = (const float*)d_in[0];
  const float* rc   = (const float*)d_in[1];
  const float* summ = (const float*)d_in[2];
  const float* mem  = (const float*)d_in[3];
  const unsigned char* amask = (const unsigned char*)d_in[4];
  const unsigned char* pmask = (const unsigned char*)d_in[5];
  const float* Wq  = (const float*)d_in[6];
  const float* bq  = (const float*)d_in[7];
  const float* Wkv = (const float*)d_in[8];
  const float* bkv = (const float*)d_in[9];
  const float* Wo  = (const float*)d_in[10];
  const float* bo  = (const float*)d_in[11];

  _Float16* qh    = (_Float16*)d_ws;
  _Float16* kh    = qh  + QH_E;
  _Float16* vhT   = kh  + QH_E;
  _Float16* attnH = vhT + QH_E;
  float* out = (float*)d_out;

  qproj_kernel<<<2624, 128, 0, stream>>>(utt, rc, summ, Wq, bq, qh);
  kvproj_kernel<<<5244, 128, 0, stream>>>(utt, rc, mem, Wkv, bkv, kh, vhT);
  pad_kernel<<<32, 256, 0, stream>>>(kh, vhT);
  // 21 q-groups (4 q-tiles each) x 128 head-batches
  attn_kernel<<<21 * 128, 128, 0, stream>>>(amask, pmask, qh, kh, vhT, attnH);
  oproj_kernel<<<2624, 128, 0, stream>>>(attnH, Wo, bo, out);
}
